// SOMVAE_1005022347586
// MI455X (gfx1250) — matrix operand order repaired and checked by reading, not compiled, not run on hardware
//
#include <hip/hip_runtime.h>
#include <math.h>

// ---------------------------------------------------------------------------
// SOM-VAE forward for MI455X (gfx1250, wave32, WMMA).
// B=131072, T=256, H=512, L=64, K=256 codes (16x16 SOM grid).
// ---------------------------------------------------------------------------

typedef __attribute__((ext_vector_type(16))) _Float16 v16h;
typedef __attribute__((ext_vector_type(8)))  _Float16 v8h;
typedef __attribute__((ext_vector_type(8)))  float    v8f;

#define B_DIM 131072
#define T_DIM 256
#define H_DIM 512
#define L_DIM 64
#define K_CODES 256

// ---------------- WMMA helpers -------------------------------------------

__device__ __forceinline__ v8f wmma_f16(v16h a, v16h b, v8f c) {
  // D = A(16x32 f16) * B(32x16 f16) + C(16x16 f32)
  return __builtin_amdgcn_wmma_f32_16x16x32_f16(
      /*neg_a=*/false, a, /*neg_b=*/false, b,
      /*c_mod=*/(short)0, c, /*reuse_a=*/false, /*reuse_b=*/false);
}

// A fragment (16x32, f16) from LDS tile laid row-major [16][ldk].
// Layout: lanes 0-15 hold K 0..15, lanes 16-31 hold K 16..31 (matches load_b_gl).
__device__ __forceinline__ v16h load_a_lds(const _Float16* base, int ldk, int k0) {
  int lane = threadIdx.x & 31;
  int m    = lane & 15;
  int kb   = (lane >> 4) << 4;                 // 0 or 16
  const _Float16* p = base + m * ldk + k0 + kb;
  v8h lo = *(const v8h*)p;
  v8h hi = *(const v8h*)(p + 8);
  v16h a;
#pragma unroll
  for (int i = 0; i < 8; ++i) { a[i] = lo[i]; a[8 + i] = hi[i]; }
  return a;
}

// B fragment (32x16, f16) from a column-major ("transposed") f16 weight,
// Wt[n][k] at Wt[n*K + k]. Lanes 0-15 hold K 0..15, lanes 16-31 hold K 16..31.
__device__ __forceinline__ v16h load_b_gl(const _Float16* Wt, int K, int n0, int k0) {
  int lane = threadIdx.x & 31;
  int n    = n0 + (lane & 15);
  int kb   = (lane >> 4) << 4;                 // 0 or 16
  return *(const v16h*)(Wt + n * K + k0 + kb); // 32B contiguous
}

// stage f32 -> f16 into LDS using 128-bit global loads
__device__ __forceinline__ void stage_f16(const float* __restrict__ src,
                                          _Float16* __restrict__ dst,
                                          int n_elems, int tid, int nthreads) {
  const float4* sv = (const float4*)src;
  for (int i = tid; i < (n_elems >> 2); i += nthreads) {
    float4 v = sv[i];
    int o = i << 2;
    dst[o + 0] = (_Float16)v.x;
    dst[o + 1] = (_Float16)v.y;
    dst[o + 2] = (_Float16)v.z;
    dst[o + 3] = (_Float16)v.w;
  }
}

// ---------------- weight prep --------------------------------------------

// W row-major [K][N] f32 -> Wt [N][K] f16
__global__ void convert_transpose(const float* __restrict__ W,
                                  _Float16* __restrict__ Wt, int K, int N) {
  int i = blockIdx.x * blockDim.x + threadIdx.x;
  if (i < K * N) {
    int k = i / N, n = i % N;
    Wt[n * K + k] = (_Float16)W[i];
  }
}

// emb [256][64] f32 -> f16 copy (already column-major for the distance GEMM's
// B matrix: column n = code n, contiguous over L) + per-code squared norm.
__global__ void convert_emb(const float* __restrict__ emb,
                            _Float16* __restrict__ embh,
                            float* __restrict__ embn) {
  int code = blockIdx.x, l = threadIdx.x;      // <<<256, 64>>>
  float v = emb[code * L_DIM + l];
  embh[code * L_DIM + l] = (_Float16)v;
  __shared__ float s[64];
  s[l] = v * v;
  __syncthreads();
#pragma unroll
  for (int st = 32; st > 0; st >>= 1) {
    if (l < st) s[l] += s[l + st];
    __syncthreads();
  }
  if (l == 0) embn[code] = s[0];
}

// ---------------- encoder: z_e = relu(relu(x W1 + b1) W2 + b2) ------------
// 32 rows/block, 8 waves. GEMM1: 64 tiles -> 8/wave (A hoisted per k-step).
// GEMM2: 8 tiles -> 1/wave (full utilization).

__global__ void encoder_kernel(const float* __restrict__ x,
                               const _Float16* __restrict__ W1t,  // [512][256]
                               const float* __restrict__ b1,
                               const _Float16* __restrict__ W2t,  // [64][512]
                               const float* __restrict__ b2,
                               float* __restrict__ z_e) {
  __shared__ __align__(32) _Float16 xs[32 * T_DIM];  // 16 KB
  __shared__ __align__(32) _Float16 hs[32 * H_DIM];  // 32 KB
  int m0  = blockIdx.x * 32;
  int tid = threadIdx.x;
  int wave = tid >> 5, lane = tid & 31;
  int mt = wave >> 2;               // M-tile (0/1): rows mt*16 .. mt*16+15
  int ng = wave & 3;                // N group
  int nl = lane & 15;
  int mo = (lane >> 4) * 8;

  stage_f16(x + (size_t)m0 * T_DIM, xs, 32 * T_DIM, tid, 256);
  __syncthreads();

  const _Float16* xtile = xs + mt * 16 * T_DIM;
  v8f zero = {};

  // GEMM1: [16,256] x [256,512]; 8 N-tiles per wave, A loaded once per k-step.
  v8f acc[8];
#pragma unroll
  for (int t = 0; t < 8; ++t) acc[t] = zero;
#pragma unroll
  for (int k0 = 0; k0 < T_DIM; k0 += 32) {
    v16h a = load_a_lds(xtile, T_DIM, k0);
#pragma unroll
    for (int t = 0; t < 8; ++t) {
      int n0 = ng * 128 + t * 16;
      acc[t] = wmma_f16(a, load_b_gl(W1t, T_DIM, n0, k0), acc[t]);
    }
  }
#pragma unroll
  for (int t = 0; t < 8; ++t) {
    int n = ng * 128 + t * 16 + nl;
    float bias = b1[n];
#pragma unroll
    for (int r = 0; r < 8; ++r) {
      float v = acc[t][r] + bias;
      hs[(mt * 16 + mo + r) * H_DIM + n] = (_Float16)(v > 0.0f ? v : 0.0f);
    }
  }
  __syncthreads();

  // GEMM2: [16,512] x [512,64]; one 16-col tile per wave.
  const _Float16* htile = hs + mt * 16 * H_DIM;
  int n0 = ng * 16;
  v8f acc2 = zero;
#pragma unroll
  for (int k0 = 0; k0 < H_DIM; k0 += 32) {
    v16h a = load_a_lds(htile, H_DIM, k0);
    acc2 = wmma_f16(a, load_b_gl(W2t, H_DIM, n0, k0), acc2);
  }
  int n = n0 + nl;
  float bias = b2[n];
#pragma unroll
  for (int r = 0; r < 8; ++r) {
    float v = acc2[r] + bias;
    z_e[(m0 + mt * 16 + mo + r) * L_DIM + n] = (v > 0.0f ? v : 0.0f);
  }
}

// ---------------- SOM: distances, argmin, quantize, neighbors -------------

__global__ void som_kernel(const float* __restrict__ z_e,
                           const float* __restrict__ emb,     // f32 [256][64]
                           const _Float16* __restrict__ embh, // f16 [256][64]
                           const float* __restrict__ embn,    // [256]
                           float* __restrict__ dist,          // [B,256]
                           float* __restrict__ kout,          // [B]
                           float* __restrict__ zq,            // [B,64]
                           float* __restrict__ neigh) {       // [B,5,64]
  __shared__ __align__(32) _Float16 zs[8][16 * L_DIM];  // 16 KB
  __shared__ float zsq[8][16];
  __shared__ int   kidx[8][16];
  int wave = threadIdx.x >> 5, lane = threadIdx.x & 31;
  int rowbase = blockIdx.x * 128 + wave * 16;

  stage_f16(z_e + (size_t)rowbase * L_DIM, zs[wave], 16 * L_DIM, lane, 32);
  if (lane < 16) {
    float s = 0.0f;
    const float4* zr = (const float4*)(z_e + (rowbase + lane) * L_DIM);
#pragma unroll
    for (int j = 0; j < L_DIM / 4; ++j) {
      float4 v = zr[j];
      s += v.x * v.x + v.y * v.y + v.z * v.z + v.w * v.w;
    }
    zsq[wave][lane] = s;
  }
  __syncthreads();

  v16h a0 = load_a_lds(zs[wave], L_DIM, 0);
  v16h a1 = load_a_lds(zs[wave], L_DIM, 32);

  float best[8]; int bidx[8];
#pragma unroll
  for (int r = 0; r < 8; ++r) { best[r] = 3.4e38f; bidx[r] = 0; }
  int mo = (lane >> 4) * 8;
  int nl = lane & 15;

  for (int t = 0; t < 16; ++t) {
    int n0 = t * 16;
    v8f acc = {};
    acc = wmma_f16(a0, load_b_gl(embh, L_DIM, n0, 0), acc);
    acc = wmma_f16(a1, load_b_gl(embh, L_DIM, n0, 32), acc);
    int n = n0 + nl;
    float en = embn[n];
#pragma unroll
    for (int r = 0; r < 8; ++r) {
      int m = mo + r;
      float d = zsq[wave][m] - 2.0f * acc[r] + en;
      dist[(rowbase + m) * K_CODES + n] = d;
      if (d < best[r]) { best[r] = d; bidx[r] = n; }
    }
  }

  // argmin across the 16 lanes of each half-wave (rows 0-7 / 8-15).
#pragma unroll
  for (int s = 1; s < 16; s <<= 1) {
#pragma unroll
    for (int r = 0; r < 8; ++r) {
      float ov = __shfl_xor(best[r], s, 32);
      int   oi = __shfl_xor(bidx[r], s, 32);
      if (ov < best[r] || (ov == best[r] && oi < bidx[r])) {
        best[r] = ov; bidx[r] = oi;
      }
    }
  }
  if (nl == 0) {
#pragma unroll
    for (int r = 0; r < 8; ++r) {
      kidx[wave][mo + r] = bidx[r];
      kout[rowbase + mo + r] = (float)bidx[r];
    }
  }
  __syncthreads();

  // cooperative gathers: z_q + 5-neighborhood (zeroed at grid boundaries)
  for (int i = 0; i < 16; ++i) {
    int k  = kidx[wave][i];
    int k1 = k >> 4, k2 = k & 15;
    int row = rowbase + i;
    for (int j = lane; j < L_DIM; j += 32) {
      float v = emb[k * L_DIM + j];
      zq[row * L_DIM + j] = v;
      neigh[(row * 5 + 0) * L_DIM + j] = v;
    }
    int  nc[4]; bool nv[4];
    nv[0] = (k1 < 15); nc[0] = (k1 + 1) * 16 + k2;  // up
    nv[1] = (k1 > 0);  nc[1] = (k1 - 1) * 16 + k2;  // down
    nv[2] = (k2 < 15); nc[2] = k1 * 16 + k2 + 1;    // right
    nv[3] = (k2 > 0);  nc[3] = k1 * 16 + k2 - 1;    // left
#pragma unroll
    for (int t = 0; t < 4; ++t) {
      for (int j = lane; j < L_DIM; j += 32) {
        neigh[(row * 5 + 1 + t) * L_DIM + j] = nv[t] ? emb[nc[t] * L_DIM + j] : 0.0f;
      }
    }
  }
}

// ---------------- decoder: sigmoid(relu(z W1 + b1) W2 + b2) ---------------
// 32 rows/block, 8 waves. GEMM1: 64 tiles -> 8/wave (A already in regs).
// GEMM2: 32 tiles -> 4/wave (A hoisted per k-step).

__global__ void decoder_kernel(const float* __restrict__ z,        // [B,64]
                               const _Float16* __restrict__ W1t,   // [512][64]
                               const float* __restrict__ b1,
                               const _Float16* __restrict__ W2t,   // [256][512]
                               const float* __restrict__ b2,
                               float* __restrict__ out) {          // [B,256]
  __shared__ __align__(32) _Float16 zsh[32 * L_DIM];   // 4 KB
  __shared__ __align__(32) _Float16 hs[32 * H_DIM];    // 32 KB
  int m0  = blockIdx.x * 32;
  int tid = threadIdx.x;
  int wave = tid >> 5, lane = tid & 31;
  int mt = wave >> 2;
  int ng = wave & 3;
  int nl = lane & 15;
  int mo = (lane >> 4) * 8;

  stage_f16(z + (size_t)m0 * L_DIM, zsh, 32 * L_DIM, tid, 256);
  __syncthreads();

  const _Float16* ztile = zsh + mt * 16 * L_DIM;
  v16h a0 = load_a_lds(ztile, L_DIM, 0);
  v16h a1 = load_a_lds(ztile, L_DIM, 32);
  v8f zero = {};

  // GEMM1: [16,64] x [64,512]; 8 N-tiles per wave, A frags held in registers.
#pragma unroll
  for (int t = 0; t < 8; ++t) {
    int n0 = ng * 128 + t * 16;
    v8f acc = zero;
    acc = wmma_f16(a0, load_b_gl(W1t, L_DIM, n0, 0), acc);
    acc = wmma_f16(a1, load_b_gl(W1t, L_DIM, n0, 32), acc);
    int n = n0 + nl;
    float bias = b1[n];
#pragma unroll
    for (int r = 0; r < 8; ++r) {
      float v = acc[r] + bias;
      hs[(mt * 16 + mo + r) * H_DIM + n] = (_Float16)(v > 0.0f ? v : 0.0f);
    }
  }
  __syncthreads();

  // GEMM2: [16,512] x [512,256]; 4 N-tiles per wave, A loaded once per k-step.
  const _Float16* htile = hs + mt * 16 * H_DIM;
  v8f acc2[4];
#pragma unroll
  for (int t = 0; t < 4; ++t) acc2[t] = zero;
#pragma unroll
  for (int k0 = 0; k0 < H_DIM; k0 += 32) {
    v16h a = load_a_lds(htile, H_DIM, k0);
#pragma unroll
    for (int t = 0; t < 4; ++t) {
      int n0 = ng * 64 + t * 16;
      acc2[t] = wmma_f16(a, load_b_gl(W2t, H_DIM, n0, k0), acc2[t]);
    }
  }
#pragma unroll
  for (int t = 0; t < 4; ++t) {
    int n = ng * 64 + t * 16 + nl;
    float bias = b2[n];
#pragma unroll
    for (int r = 0; r < 8; ++r) {
      float v = acc2[t][r] + bias;
      out[(m0 + mt * 16 + mo + r) * T_DIM + n] = 1.0f / (1.0f + expf(-v));
    }
  }
}

// ---------------- host launch ---------------------------------------------

extern "C" void kernel_launch(void* const* d_in, const int* in_sizes, int n_in,
                              void* d_out, int out_size, void* d_ws, size_t ws_size,
                              hipStream_t stream) {
  const float* x    = (const float*)d_in[0];
  const float* emb  = (const float*)d_in[1];
  const float* W_e1 = (const float*)d_in[2];
  const float* b_e1 = (const float*)d_in[3];
  const float* W_e2 = (const float*)d_in[4];
  const float* b_e2 = (const float*)d_in[5];
  const float* Wq1  = (const float*)d_in[6];
  const float* bq1  = (const float*)d_in[7];
  const float* Wq2  = (const float*)d_in[8];
  const float* bq2  = (const float*)d_in[9];
  const float* We1  = (const float*)d_in[10];
  const float* be1  = (const float*)d_in[11];
  const float* We2  = (const float*)d_in[12];
  const float* be2  = (const float*)d_in[13];

  // workspace: transposed f16 weights + f16 codebook + code norms (~1 MB)
  _Float16* wsh   = (_Float16*)d_ws;
  _Float16* W_e1t = wsh;                  // [512][256]
  _Float16* W_e2t = wsh + 131072;         // [64][512]
  _Float16* Wq1t  = wsh + 163840;         // [512][64]
  _Float16* Wq2t  = wsh + 196608;         // [256][512]
  _Float16* We1t  = wsh + 327680;         // [512][64]
  _Float16* We2t  = wsh + 360448;         // [256][512]
  _Float16* embh  = wsh + 491520;         // [256][64]
  float*    embn  = (float*)(wsh + 507904);

  // d_out: x_hat_q, x_hat_e, z_e, z_q, k, z_dist_flat, z_q_neighbors (flat f32)
  float* out  = (float*)d_out;
  float* xq   = out;
  float* xe   = out + (size_t)B_DIM * T_DIM;
  float* ze   = out + (size_t)2 * B_DIM * T_DIM;
  float* zq   = ze + (size_t)B_DIM * L_DIM;
  float* kf   = zq + (size_t)B_DIM * L_DIM;
  float* dist = kf + (size_t)B_DIM;
  float* neigh= dist + (size_t)B_DIM * K_CODES;

  // weight prep
  convert_transpose<<<(T_DIM * H_DIM + 255) / 256, 256, 0, stream>>>(W_e1, W_e1t, T_DIM, H_DIM);
  convert_transpose<<<(H_DIM * L_DIM + 255) / 256, 256, 0, stream>>>(W_e2, W_e2t, H_DIM, L_DIM);
  convert_transpose<<<(L_DIM * H_DIM + 255) / 256, 256, 0, stream>>>(Wq1, Wq1t, L_DIM, H_DIM);
  convert_transpose<<<(H_DIM * T_DIM + 255) / 256, 256, 0, stream>>>(Wq2, Wq2t, H_DIM, T_DIM);
  convert_transpose<<<(L_DIM * H_DIM + 255) / 256, 256, 0, stream>>>(We1, We1t, L_DIM, H_DIM);
  convert_transpose<<<(H_DIM * T_DIM + 255) / 256, 256, 0, stream>>>(We2, We2t, H_DIM, T_DIM);
  convert_emb<<<K_CODES, 64, 0, stream>>>(emb, embh, embn);

  // pipeline
  encoder_kernel<<<B_DIM / 32, 256, 0, stream>>>(x, W_e1t, b_e1, W_e2t, b_e2, ze);
  som_kernel<<<B_DIM / 128, 256, 0, stream>>>(ze, emb, embh, embn, dist, kf, zq, neigh);
  decoder_kernel<<<B_DIM / 32, 256, 0, stream>>>(zq, Wq1t, bq1, Wq2t, bq2, xq);
  decoder_kernel<<<B_DIM / 32, 256, 0, stream>>>(ze, We1t, be1, We2t, be2, xe);
}
